// UncertaintyGuidedAttention_54142357733943
// MI455X (gfx1250) — compile-verified, matrix operand order checked by reading, not run on hardware
//
#include <hip/hip_runtime.h>
#include <hip/hip_bf16.h>
#include <math.h>
#include <stdint.h>

// ---------------------------------------------------------------------------
// UncertaintyGuidedAttention for MI455X (gfx1250, wave32, WMMA bf16 path)
// B=2, N=2048, C=1024, H=16, D=64
// Attention: transposed score/output formulation + double-buffered async
// global->LDS staging of K/V tiles shared by all 4 waves of a block.
// ---------------------------------------------------------------------------

#define NB 2
#define NN 2048
#define NC 1024
#define NH 16
#define ND 64
#define BN (NB * NN)   // 4096 rows
#define LDS_LD 72      // padded LDS row stride (elements) to spread banks

typedef __attribute__((ext_vector_type(16))) __bf16       v16bf;
typedef __attribute__((ext_vector_type(8)))  float        v8f;
typedef __attribute__((ext_vector_type(4)))  unsigned int u32x4;
typedef __attribute__((ext_vector_type(4)))  int          i32x4;

#if defined(__has_builtin)
#if __has_builtin(__builtin_amdgcn_global_load_async_to_lds_b128)
#define HAVE_ASYNC_LDS 1
#endif
#endif
#ifndef HAVE_ASYNC_LDS
#define HAVE_ASYNC_LDS 0
#endif

#define AS1 __attribute__((address_space(1)))
#define AS3 __attribute__((address_space(3)))

// one 16-byte async global->LDS transfer (ASYNCcnt-tracked on gfx1250)
// builtin signature (from hipcc diagnostic): (V4i as1*, V4i as3*, imm, imm)
__device__ __forceinline__ void async_copy_b128(const void* g, void* l) {
#if HAVE_ASYNC_LDS
  __builtin_amdgcn_global_load_async_to_lds_b128(
      (AS1 i32x4*)(uintptr_t)g, (AS3 i32x4*)(uintptr_t)l, 0, 0);
#else
  *(u32x4*)l = *(const u32x4*)g;
#endif
}

__device__ __forceinline__ void wait_async_le8() {
#if HAVE_ASYNC_LDS
#if __has_builtin(__builtin_amdgcn_s_wait_asynccnt)
  __builtin_amdgcn_s_wait_asynccnt(8);
#else
  asm volatile("s_wait_asynccnt 0x8" ::: "memory");
#endif
#endif
}
__device__ __forceinline__ void wait_async_le0() {
#if HAVE_ASYNC_LDS
#if __has_builtin(__builtin_amdgcn_s_wait_asynccnt)
  __builtin_amdgcn_s_wait_asynccnt(0);
#else
  asm volatile("s_wait_asynccnt 0x0" ::: "memory");
#endif
#endif
}

// round-to-nearest-even float -> bf16 bit pattern
__device__ __forceinline__ unsigned short f2bf(float f) {
  union { float f; unsigned u; } c; c.f = f;
  unsigned u = c.u;
  unsigned r = (u + 0x7FFFu + ((u >> 16) & 1u)) >> 16;
  return (unsigned short)r;
}

// Load one 16x32 bf16 A-fragment (or B^T-fragment) for wave32 WMMA.
// Layout (ISA 7.12.2): lanes 0-15 hold K = [0..7] and [16..23]; lanes 16-31
// hold K = [8..15] and [24..31]. `row` must already include (lane & 15).
__device__ __forceinline__ v16bf load_frag(const unsigned short* __restrict__ p,
                                           int row, int ld, int k0, int half) {
  const unsigned short* q = p + (size_t)row * ld + k0 + half * 8;
  union { v16bf v; u32x4 q4[2]; } u;
  u.q4[0] = *(const u32x4*)(q);
  u.q4[1] = *(const u32x4*)(q + 16);
  return u.v;
}

__device__ __forceinline__ v8f wmma_bf16(v16bf a, v16bf b, v8f c) {
  // D = A(16x32) * B(32x16) + C, fp32 accumulate
  return __builtin_amdgcn_wmma_f32_16x16x32_bf16(
      false, a, false, b, (short)0, c, false, false);
}

// ---------------------------------------------------------------------------
// Kernel 1a: fp32 -> bf16 elementwise (for x)
// ---------------------------------------------------------------------------
__global__ void cvt_f32_bf16_kernel(const float* __restrict__ src,
                                    unsigned short* __restrict__ dst, int n) {
  int i = blockIdx.x * blockDim.x + threadIdx.x;
  if (i < n) dst[i] = f2bf(src[i]);
}

// ---------------------------------------------------------------------------
// Kernel 1b: W (K x N fp32 row-major) -> W^T (N x K bf16 row-major)
// ---------------------------------------------------------------------------
__global__ void cvt_transpose_kernel(const float* __restrict__ W,
                                     unsigned short* __restrict__ Wt) {
  __shared__ float tile[32][33];
  int k0 = blockIdx.y * 32, n0 = blockIdx.x * 32;
  int tx = threadIdx.x, ty = threadIdx.y;
  for (int j = ty; j < 32; j += 8)
    tile[j][tx] = W[(size_t)(k0 + j) * NC + n0 + tx];
  __syncthreads();
  for (int j = ty; j < 32; j += 8)
    Wt[(size_t)(n0 + j) * NC + k0 + tx] = f2bf(tile[tx][j]);
}

// ---------------------------------------------------------------------------
// Kernel 2/4: WMMA GEMM  out(4096 x 1024) = A(4096 x 1024) * B(1024 x 1024) + bias
// MODE 0: bf16 (B,H,N,D); MODE 1: bf16 (B,H,D,N); MODE 2: fp32 (B,N,C)
// grid (4, 128), block 128 (4 waves); each wave computes a 32x64 output tile.
// ---------------------------------------------------------------------------
template <int MODE>
__global__ void wmma_gemm_kernel(const unsigned short* __restrict__ Abf,
                                 const unsigned short* __restrict__ Btbf,
                                 const float* __restrict__ bias,
                                 void* __restrict__ out) {
  const int wave  = threadIdx.x >> 5;
  const int lane  = threadIdx.x & 31;
  const int col   = lane & 15;
  const int half  = lane >> 4;
  const int mbase = blockIdx.y * 32;
  const int nbase = (blockIdx.x * 4 + wave) * 64;

  v8f acc[2][4];
  #pragma unroll
  for (int t = 0; t < 2; ++t)
    #pragma unroll
    for (int s = 0; s < 4; ++s) acc[t][s] = (v8f)(0.0f);

  for (int k0 = 0; k0 < NC; k0 += 32) {
    const v16bf a0 = load_frag(Abf, mbase + col,      NC, k0, half);
    const v16bf a1 = load_frag(Abf, mbase + 16 + col, NC, k0, half);
    #pragma unroll
    for (int s = 0; s < 4; ++s) {
      const v16bf b = load_frag(Btbf, nbase + s * 16 + col, NC, k0, half);
      acc[0][s] = wmma_bf16(a0, b, acc[0][s]);
      acc[1][s] = wmma_bf16(a1, b, acc[1][s]);
    }
  }

  #pragma unroll
  for (int t = 0; t < 2; ++t) {
    #pragma unroll
    for (int s = 0; s < 4; ++s) {
      const int   c  = nbase + s * 16 + col;
      const float bv = bias[c];
      #pragma unroll
      for (int j = 0; j < 8; ++j) {
        const int   r = mbase + t * 16 + half * 8 + j;  // global row in [0, 4096)
        const float v = acc[t][s][j] + bv;
        const int b = r >> 11;
        const int n = r & (NN - 1);
        const int h = c >> 6;
        const int d = c & (ND - 1);
        if (MODE == 0) {
          ((unsigned short*)out)[(((size_t)b * NH + h) * NN + n) * ND + d] = f2bf(v);
        } else if (MODE == 1) {
          ((unsigned short*)out)[(((size_t)b * NH + h) * ND + d) * NN + n] = f2bf(v);
        } else {
          ((float*)out)[(size_t)r * NC + c] = v;
        }
      }
    }
  }
}

// ---------------------------------------------------------------------------
// Kernel 3: fused flash-attention with per-query uncertainty weighting.
// grid (B*H = 32, 32), block 128 (4 waves; each wave owns one 16-query tile,
// all 4 share the same (b,h) K/V stream).
//
// Per 64-key block the K tile (64x64 bf16) and V^T tile (64x64 bf16) are
// staged ONCE into LDS with double-buffered async global->LDS copies
// (ASYNCcnt), so the 4 waves read fragments from LDS instead of 4x-redundant
// global loads, and the next tile's DMA overlaps this tile's WMMAs.
// ---------------------------------------------------------------------------
__global__ void attn_flash_kernel(const unsigned short* __restrict__ Q,
                                  const unsigned short* __restrict__ K,
                                  const unsigned short* __restrict__ Vt,
                                  const float* __restrict__ unc,
                                  unsigned short* __restrict__ O) {
  __shared__ __align__(16) unsigned short kTile[2][64 * LDS_LD];
  __shared__ __align__(16) unsigned short vTile[2][64 * LDS_LD];

  const int tid   = threadIdx.x;
  const int wave  = tid >> 5;
  const int lane  = tid & 31;
  const int col   = lane & 15;
  const int half  = lane >> 4;
  const int bh    = blockIdx.x;
  const int b     = bh >> 4;
  const int h     = bh & 15;
  const int qbase = (blockIdx.y * 4 + wave) * 16;

  const unsigned short* Qb = Q  + (size_t)bh * NN * ND;
  const unsigned short* Kb = K  + (size_t)bh * NN * ND;
  const unsigned short* Vb = Vt + (size_t)bh * ND * NN;

  // cooperative staging assignment: 2 threads per 64-element row
  const int crow = tid >> 1;            // 0..63
  const int coff = (tid & 1) * 32;      // element offset within row

  // Q^T fragments (B operand of the score WMMA) stay resident in registers.
  const v16bf qb0 = load_frag(Qb, qbase + col, ND, 0,  half);
  const v16bf qb1 = load_frag(Qb, qbase + col, ND, 32, half);

  // per-lane (per-query) combined factor: scale / (1 + uncertainty)
  const float fw = 0.125f / (1.0f + unc[(size_t)b * NN + qbase + col]);
  float mrow = -1e30f;
  float lrow = 0.0f;

  v8f acc[4];  // out^T: rows = head-dim (4 subtiles of 16), cols = queries
  #pragma unroll
  for (int s = 0; s < 4; ++s) acc[s] = (v8f)(0.0f);

  // stage one 64-key K + V^T tile pair into LDS buffer `buf`
  auto stage = [&](int buf, int kb) {
    const unsigned short* gk = Kb + ((size_t)kb + crow) * ND + coff;
    const unsigned short* gv = Vb + (size_t)crow * NN + kb + coff;
    unsigned short* lk = &kTile[buf][crow * LDS_LD + coff];
    unsigned short* lv = &vTile[buf][crow * LDS_LD + coff];
    #pragma unroll
    for (int c = 0; c < 4; ++c) {
      async_copy_b128(gk + c * 8, lk + c * 8);
      async_copy_b128(gv + c * 8, lv + c * 8);
    }
  };

#if HAVE_ASYNC_LDS
  stage(0, 0);
  int buf = 0;
#endif
  for (int kb = 0; kb < NN; kb += 64) {
#if HAVE_ASYNC_LDS
    const bool more = (kb + 64) < NN;
    if (more) {
      stage(buf ^ 1, kb + 64);   // DMA next tile while we compute this one
      wait_async_le8();          // previous batch (this tile) fully landed
    } else {
      wait_async_le0();
    }
    __syncthreads();
    const unsigned short* kT = &kTile[buf][0];
    const unsigned short* vT = &vTile[buf][0];
#else
    __syncthreads();
    stage(0, kb);
    __syncthreads();
    const unsigned short* kT = &kTile[0][0];
    const unsigned short* vT = &vTile[0][0];
#endif
    // warm L2 for the tile after next
    __builtin_prefetch(Kb + (size_t)(kb + 128 + crow) * ND + coff, 0, 2);
    __builtin_prefetch(Vb + (size_t)crow * NN + kb + 128 + coff, 0, 2);

    // ---- transposed scores from LDS: st[t] covers local keys [16t,16t+16) ----
    v8f st[4];
    #pragma unroll
    for (int t = 0; t < 4; ++t) {
      v8f s = (v8f)(0.0f);
      s = wmma_bf16(load_frag(kT, t * 16 + col, LDS_LD, 0,  half), qb0, s);
      s = wmma_bf16(load_frag(kT, t * 16 + col, LDS_LD, 32, half), qb1, s);
      st[t] = s;
    }

    // ---- per-lane online softmax over 32 in-lane scores (+1 shuffle) ----
    float m_blk = -1e30f;
    #pragma unroll
    for (int t = 0; t < 4; ++t)
      #pragma unroll
      for (int j = 0; j < 8; ++j) {
        st[t][j] *= fw;
        m_blk = fmaxf(m_blk, st[t][j]);
      }
    m_blk = fmaxf(m_blk, __shfl_xor(m_blk, 16, 32));
    const float mnew  = fmaxf(mrow, m_blk);
    const float alpha = __expf(mrow - mnew);
    mrow = mnew;

    // exp + pack straight into the two PV B-fragments (layout-coincident)
    union { v16bf v; unsigned short s[16]; } pa[2];
    float rs = 0.0f;
    #pragma unroll
    for (int t = 0; t < 4; ++t)
      #pragma unroll
      for (int j = 0; j < 8; ++j) {
        const float e = __expf(st[t][j] - mnew);
        rs += e;
        pa[t >> 1].s[(t & 1) * 8 + j] = f2bf(e);
      }
    rs += __shfl_xor(rs, 16, 32);
    lrow = lrow * alpha + rs;

    #pragma unroll
    for (int s = 0; s < 4; ++s) acc[s] = acc[s] * alpha;  // per-query rescale

    // ---- out^T += V^T(16d x 32k) * P^T(32k x 16q) from LDS ----
    #pragma unroll
    for (int s = 0; s < 4; ++s) {
      acc[s] = wmma_bf16(load_frag(vT, s * 16 + col, LDS_LD, 0,  half), pa[0].v, acc[s]);
      acc[s] = wmma_bf16(load_frag(vT, s * 16 + col, LDS_LD, 32, half), pa[1].v, acc[s]);
    }
    __syncthreads();   // all waves done with this buffer before it is reused
#if HAVE_ASYNC_LDS
    buf ^= 1;
#endif
  }

  // ---- normalize and store out^T to O as (B, N, H*D) = (B, N, C) ----
  const float inv = 1.0f / lrow;
  const size_t orow = ((size_t)b * NN + qbase + col) * NC + (size_t)h * ND;
  #pragma unroll
  for (int s = 0; s < 4; ++s)
    #pragma unroll
    for (int j = 0; j < 8; ++j) {
      const int d = s * 16 + half * 8 + j;
      O[orow + d] = f2bf(acc[s][j] * inv);
    }
}

// ---------------------------------------------------------------------------
// Host-side launcher
// ---------------------------------------------------------------------------
extern "C" void kernel_launch(void* const* d_in, const int* in_sizes, int n_in,
                              void* d_out, int out_size, void* d_ws, size_t ws_size,
                              hipStream_t stream) {
  const float* x   = (const float*)d_in[0];
  const float* unc = (const float*)d_in[1];
  const float* Wq  = (const float*)d_in[2];
  const float* bq  = (const float*)d_in[3];
  const float* Wk  = (const float*)d_in[4];
  const float* bk  = (const float*)d_in[5];
  const float* Wv  = (const float*)d_in[6];
  const float* bv  = (const float*)d_in[7];
  const float* Wo  = (const float*)d_in[8];
  const float* bo  = (const float*)d_in[9];

  // workspace layout (ushort elements)
  unsigned short* ws  = (unsigned short*)d_ws;
  unsigned short* xbf = ws;                     // 4096*1024
  unsigned short* wqT = xbf + (size_t)BN * NC;  // 1024*1024
  unsigned short* wkT = wqT + (size_t)NC * NC;
  unsigned short* wvT = wkT + (size_t)NC * NC;
  unsigned short* woT = wvT + (size_t)NC * NC;
  unsigned short* Qb  = woT + (size_t)NC * NC;  // (B,H,N,D)
  unsigned short* Kb  = Qb  + (size_t)BN * NC;
  unsigned short* Vtb = Kb  + (size_t)BN * NC;  // (B,H,D,N)
  unsigned short* Ob  = Vtb + (size_t)BN * NC;  // (B,N,C)

  // 1) precision conversion + weight transposes
  cvt_f32_bf16_kernel<<<(BN * NC) / 256, 256, 0, stream>>>(x, xbf, BN * NC);
  dim3 tg(NC / 32, NC / 32), tb(32, 8);
  cvt_transpose_kernel<<<tg, tb, 0, stream>>>(Wq, wqT);
  cvt_transpose_kernel<<<tg, tb, 0, stream>>>(Wk, wkT);
  cvt_transpose_kernel<<<tg, tb, 0, stream>>>(Wv, wvT);
  cvt_transpose_kernel<<<tg, tb, 0, stream>>>(Wo, woT);

  // 2) QKV projections (WMMA)
  dim3 gg(4, BN / 32);
  wmma_gemm_kernel<0><<<gg, 128, 0, stream>>>(xbf, wqT, bq, (void*)Qb);
  wmma_gemm_kernel<0><<<gg, 128, 0, stream>>>(xbf, wkT, bk, (void*)Kb);
  wmma_gemm_kernel<1><<<gg, 128, 0, stream>>>(xbf, wvT, bv, (void*)Vtb);

  // 3) fused uncertainty-weighted flash attention (WMMA + async LDS staging)
  attn_flash_kernel<<<dim3(NB * NH, NN / 64), 128, 0, stream>>>(Qb, Kb, Vtb, unc, Ob);

  // 4) output projection (WMMA), fp32 result
  wmma_gemm_kernel<2><<<gg, 128, 0, stream>>>(Ob, woT, bo, d_out);
}
